// mLSTMCell1900_58978490909175
// MI455X (gfx1250) — matrix operand order repaired and checked by reading, not compiled, not run on hardware
//
#include <hip/hip_runtime.h>

typedef __attribute__((ext_vector_type(16))) _Float16 v16h;
typedef __attribute__((ext_vector_type(8)))  float    v8f;

#define B_ROWS 8192
#define D_IN   10
#define Dp     32      // D_IN padded to one WMMA K-step
#define H      1900
#define Hp     1920    // H padded to multiple of 32 (K dim) / 16 (N dim)
#define N4     7600    // 4*H
#define N4p    7616    // rounded up to 32

// ---------------- WMMA fragment loaders (f16, 16x16x32) ----------------
// A (16x32, MxK): lane 0-15 -> M=lane, K=0..7 (v0-3) & 16..23 (v4-7);
//                 lane 16-31 -> M=lane-16, K=8..15 & 24..31.
// Row-major source, row stride ld (multiple of 8) -> two aligned b128 loads.
__device__ __forceinline__ v16h load_frag_a(const _Float16* __restrict__ base,
                                            int row, int ld, int k0, int lane) {
  const int hf = lane >> 4;
  const _Float16* p = base + (size_t)row * ld + k0 + hf * 8;
  union { uint4 q[2]; v16h h; } u;
  u.q[0] = *(const uint4*)(p);
  u.q[1] = *(const uint4*)(p + 16);
  return u.h;
}
// B (32x16, KxN): lane 0-15 -> N=lane, K=0..7 & 16..23; lane 16-31 -> K=8..15 & 24..31.
// Source stored N-major: WT[n*ld + k] -> two aligned b128 loads.
__device__ __forceinline__ v16h load_frag_b(const _Float16* __restrict__ wt,
                                            int col, int ld, int k0, int lane) {
  const int hf = lane >> 4;
  const _Float16* p = wt + (size_t)col * ld + k0 + hf * 8;
  union { uint4 q[2]; v16h h; } u;
  u.q[0] = *(const uint4*)(p);
  u.q[1] = *(const uint4*)(p + 16);
  return u.h;
}

__device__ __forceinline__ v8f wmma_f16(v16h a, v16h b, v8f c) {
  return __builtin_amdgcn_wmma_f32_16x16x32_f16(false, a, false, b, (short)0, c,
                                                false, false);
}

// ---------------- Preprocessing kernels ----------------
// Per-column sum of squares. Block = 256 (= 4 row-groups x 64 cols), coalesced.
__global__ __launch_bounds__(256) void colnorm_kernel(
    const float* __restrict__ W, int K, int N, float* __restrict__ ss) {
  const int tid = threadIdx.x;
  const int cn = tid & 63;
  const int rg = tid >> 6;
  const int n = blockIdx.x * 64 + cn;
  float p = 0.f;
  if (n < N) {
    for (int k = rg; k < K; k += 4) {
      float v = W[(size_t)k * N + n];
      p += v * v;
    }
  }
  __shared__ float s[256];
  s[tid] = p;
  __syncthreads();
  if (rg == 0 && n < N)
    ss[n] = s[cn] + s[64 + cn] + s[128 + cn] + s[192 + cn];
}

// Fused: normalize column n (gain * rsqrt(max(ss,1e-12))), transpose to
// N-major, convert to f16, zero-pad to [Nr x Kp]. 16x16 LDS tiles, coalesced.
__global__ __launch_bounds__(256) void transpose_scale_kernel(
    const float* __restrict__ W, const float* __restrict__ ss,
    const float* __restrict__ g, int K, int N, int Kp,
    _Float16* __restrict__ WT) {
  __shared__ float tile[16][17];
  const int tx = threadIdx.x & 15;
  const int ty = threadIdx.x >> 4;
  const int nIn = blockIdx.y * 16 + tx;
  const int kIn = blockIdx.x * 16 + ty;
  float v = 0.f;
  if (nIn < N && kIn < K) {
    float sc = g[nIn] * rsqrtf(fmaxf(ss[nIn], 1e-12f));
    v = W[(size_t)kIn * N + nIn] * sc;
  }
  tile[ty][tx] = v;
  __syncthreads();
  const int kOut = blockIdx.x * 16 + tx;
  const int nOut = blockIdx.y * 16 + ty;
  WT[(size_t)nOut * Kp + kOut] = (_Float16)tile[tx][ty];
}

// f32 [rows,C] -> f16 [rows,Cp], zero-padded columns.
__global__ __launch_bounds__(256) void cvt_pad_kernel(
    const float* __restrict__ X, int C, int Cp, _Float16* __restrict__ X16,
    int total) {
  int i = blockIdx.x * 256 + threadIdx.x;
  if (i >= total) return;
  int r = i / Cp;
  int c = i - r * Cp;
  X16[i] = (c < C) ? (_Float16)X[(size_t)r * C + c] : (_Float16)0.f;
}

// ---------------- Stage 1: m = (x @ wmxT) * (h @ wmhT), f16 out ----------------
// Block 128 (4 waves); wave computes 32 rows x 16 cols. Pad cols auto-zero.
__global__ __launch_bounds__(128) void gemm_m_kernel(
    const _Float16* __restrict__ h16, const _Float16* __restrict__ x16,
    const _Float16* __restrict__ wmhT, const _Float16* __restrict__ wmxT,
    _Float16* __restrict__ m16) {
  const int lane = threadIdx.x & 31;
  const int wave = threadIdx.x >> 5;
  const int hf = lane >> 4;
  const int r15 = lane & 15;
  const int rowBase = blockIdx.x * 128 + wave * 32;
  const int bcol = blockIdx.y * 16 + r15;

  v8f acc0 = {}, acc1 = {};
  for (int k0 = 0; k0 < Hp; k0 += 32) {
    v16h bfrag = load_frag_b(wmhT, bcol, Hp, k0, lane);
    v16h a0 = load_frag_a(h16, rowBase + r15, Hp, k0, lane);
    v16h a1 = load_frag_a(h16, rowBase + 16 + r15, Hp, k0, lane);
    acc0 = wmma_f16(a0, bfrag, acc0);
    acc1 = wmma_f16(a1, bfrag, acc1);
  }
  v8f accm0 = {}, accm1 = {};
  {
    v16h bx = load_frag_b(wmxT, bcol, Dp, 0, lane);
    v16h ax0 = load_frag_a(x16, rowBase + r15, Dp, 0, lane);
    v16h ax1 = load_frag_a(x16, rowBase + 16 + r15, Dp, 0, lane);
    accm0 = wmma_f16(ax0, bx, accm0);
    accm1 = wmma_f16(ax1, bx, accm1);
  }
#pragma unroll
  for (int e = 0; e < 8; ++e) {
    int m0 = rowBase + e + hf * 8;
    m16[(size_t)m0 * Hp + bcol] = (_Float16)(acc0[e] * accm0[e]);
    m16[(size_t)(m0 + 16) * Hp + bcol] = (_Float16)(acc1[e] * accm1[e]);
  }
}

// ---------------- Stage 2: z = x@wxT + m@whT + b, fused gates ----------------
// Wave: 32 rows x 16 gate-cols x 4 gates (8 f32 accum tiles = 64 VGPRs), so
// i/f/o/u for the same (row,col) are register-resident -> fused c/h epilogue.
__global__ __launch_bounds__(128) void gemm_z_kernel(
    const _Float16* __restrict__ m16, const _Float16* __restrict__ x16,
    const _Float16* __restrict__ whT, const _Float16* __restrict__ wxT,
    const float* __restrict__ bias, const float* __restrict__ c_prev,
    float* __restrict__ out) {
  const int lane = threadIdx.x & 31;
  const int wave = threadIdx.x >> 5;
  const int hf = lane >> 4;
  const int r15 = lane & 15;
  const int rowBase = blockIdx.x * 128 + wave * 32;
  const int col = blockIdx.y * 16 + r15;
  const bool valid = col < H;

  v8f acc[2][4] = {};
  for (int k0 = 0; k0 < Hp; k0 += 32) {
    v16h a0 = load_frag_a(m16, rowBase + r15, Hp, k0, lane);
    v16h a1 = load_frag_a(m16, rowBase + 16 + r15, Hp, k0, lane);
#pragma unroll
    for (int g = 0; g < 4; ++g) {
      v16h bf = load_frag_b(whT, g * H + col, Hp, k0, lane);
      acc[0][g] = wmma_f16(a0, bf, acc[0][g]);
      acc[1][g] = wmma_f16(a1, bf, acc[1][g]);
    }
  }
  {
    v16h ax0 = load_frag_a(x16, rowBase + r15, Dp, 0, lane);
    v16h ax1 = load_frag_a(x16, rowBase + 16 + r15, Dp, 0, lane);
#pragma unroll
    for (int g = 0; g < 4; ++g) {
      v16h bx = load_frag_b(wxT, g * H + col, Dp, 0, lane);
      acc[0][g] = wmma_f16(ax0, bx, acc[0][g]);
      acc[1][g] = wmma_f16(ax1, bx, acc[1][g]);
    }
  }
  float bi = 0.f, bfv = 0.f, bo = 0.f, bu = 0.f;
  if (valid) {
    bi = bias[col];
    bfv = bias[H + col];
    bo = bias[2 * H + col];
    bu = bias[3 * H + col];
  }
#pragma unroll
  for (int t = 0; t < 2; ++t) {
#pragma unroll
    for (int e = 0; e < 8; ++e) {
      int row = rowBase + t * 16 + e + hf * 8;
      float ig = 1.f / (1.f + __expf(-(acc[t][0][e] + bi)));
      float fg = 1.f / (1.f + __expf(-(acc[t][1][e] + bfv)));
      float og = 1.f / (1.f + __expf(-(acc[t][2][e] + bo)));
      float ug = tanhf(acc[t][3][e] + bu);
      float cp = valid ? c_prev[(size_t)row * H + col] : 0.f;
      float cn = fg * cp + ig * ug;
      float hn = og * tanhf(cn);
      if (valid) {
        out[(size_t)row * H + col] = hn;                      // h
        out[(size_t)B_ROWS * H + (size_t)row * H + col] = cn; // c
      }
    }
  }
}

// ---------------- Launch ----------------
extern "C" void kernel_launch(void* const* d_in, const int* in_sizes, int n_in,
                              void* d_out, int out_size, void* d_ws,
                              size_t ws_size, hipStream_t stream) {
  (void)in_sizes; (void)n_in; (void)out_size; (void)ws_size;
  const float* inputs = (const float*)d_in[0];
  const float* c_prev = (const float*)d_in[1];
  const float* h_prev = (const float*)d_in[2];
  const float* wx = (const float*)d_in[3];
  const float* wh = (const float*)d_in[4];
  const float* wmx = (const float*)d_in[5];
  const float* wmh = (const float*)d_in[6];
  const float* bias = (const float*)d_in[7];
  const float* gx = (const float*)d_in[8];
  const float* gh = (const float*)d_in[9];
  const float* gmx = (const float*)d_in[10];
  const float* gmh = (const float*)d_in[11];
  float* out = (float*)d_out;

  char* ws = (char*)d_ws;
  size_t off = 0;
  auto alloc = [&](size_t bytes) -> void* {
    void* p = ws + off;
    off += (bytes + 255) & ~(size_t)255;
    return p;
  };
  _Float16* x16 = (_Float16*)alloc((size_t)B_ROWS * Dp * 2);
  _Float16* h16 = (_Float16*)alloc((size_t)B_ROWS * Hp * 2);
  _Float16* m16 = (_Float16*)alloc((size_t)B_ROWS * Hp * 2);
  _Float16* whT = (_Float16*)alloc((size_t)N4p * Hp * 2);
  _Float16* wmhT = (_Float16*)alloc((size_t)Hp * Hp * 2);
  _Float16* wxT = (_Float16*)alloc((size_t)N4p * Dp * 2);
  _Float16* wmxT = (_Float16*)alloc((size_t)Hp * Dp * 2);
  float* ss = (float*)alloc((size_t)N4 * 4); // reused sequentially (in-order stream)

  int tot = B_ROWS * Dp;
  cvt_pad_kernel<<<(tot + 255) / 256, 256, 0, stream>>>(inputs, D_IN, Dp, x16, tot);
  tot = B_ROWS * Hp;
  cvt_pad_kernel<<<(tot + 255) / 256, 256, 0, stream>>>(h_prev, H, Hp, h16, tot);

  colnorm_kernel<<<(N4 + 63) / 64, 256, 0, stream>>>(wx, D_IN, N4, ss);
  transpose_scale_kernel<<<dim3(Dp / 16, N4p / 16), 256, 0, stream>>>(
      wx, ss, gx, D_IN, N4, Dp, wxT);
  colnorm_kernel<<<(N4 + 63) / 64, 256, 0, stream>>>(wh, H, N4, ss);
  transpose_scale_kernel<<<dim3(Hp / 16, N4p / 16), 256, 0, stream>>>(
      wh, ss, gh, H, N4, Hp, whT);
  colnorm_kernel<<<(H + 63) / 64, 256, 0, stream>>>(wmx, D_IN, H, ss);
  transpose_scale_kernel<<<dim3(Dp / 16, Hp / 16), 256, 0, stream>>>(
      wmx, ss, gmx, D_IN, H, Dp, wmxT);
  colnorm_kernel<<<(H + 63) / 64, 256, 0, stream>>>(wmh, H, H, ss);
  transpose_scale_kernel<<<dim3(Hp / 16, Hp / 16), 256, 0, stream>>>(
      wmh, ss, gmh, H, H, Hp, wmhT);

  gemm_m_kernel<<<dim3(B_ROWS / 128, Hp / 16), 128, 0, stream>>>(
      h16, x16, wmhT, wmxT, m16);
  gemm_z_kernel<<<dim3(B_ROWS / 128, (H + 15) / 16), 128, 0, stream>>>(
      m16, x16, whT, wxT, bias, c_prev, out);
}
